// GATAttention_22419729285546
// MI455X (gfx1250) — matrix operand order repaired and checked by reading, not compiled, not run on hardware
//
#include <hip/hip_runtime.h>
#include <hip/hip_bf16.h>

// ---------------------------------------------------------------------------
// GAT-style attention, fused for CDNA5 (gfx1250, wave32, WMMA 16x16x32 bf16).
//
//   q = (qs@Wq+b)/sqrt(32) -> [N,4,32]   k -> [L,4,32]   v -> [L,4,32]
//   s[n,l,h] = q[n,h,:]·k[l,h,:]          (scale pre-folded into q)
//   w = softmax over h (4 values, elementwise per (n,l))
//   out[n,h,d] = sum_l w[n,l,h] * v[l,h,d]
//
// Transposed formulation so no LDS transpose is ever needed:
//   S^T[16l x 16n] = Ktile(16l x 32dk) x Q^T(32dk x 16n)     (wmma #1)
//   O^T[16d x 16n] += V^T(16d x 32l)  x W^T(32l x 16n)       (wmma #2)
// W^T B-fragment built from S^T C-layout with ds_swizzle SWAPX16 only.
// ---------------------------------------------------------------------------

typedef __bf16 v16bf __attribute__((ext_vector_type(16)));
typedef float  v8f   __attribute__((ext_vector_type(8)));
typedef unsigned int u32x4 __attribute__((ext_vector_type(4)));
typedef float        f32x4 __attribute__((ext_vector_type(4)));

#define NN 8192
#define LL 8192

union Frag16 {           // 16 bf16 = 8 VGPRs : one WMMA A or B operand
    v16bf v;
    u32x4 q[2];
    unsigned int u[8];
};

// f32 pair -> packed bf16x2 dword, single VALU op (RNE in hardware)
static __device__ __forceinline__ unsigned int pk_bf16(float a, float b) {
    unsigned int r;
    asm("v_cvt_pk_bf16_f32 %0, %1, %2" : "=v"(r) : "v"(a), "v"(b));
    return r;
}
static __device__ __forceinline__ unsigned short f2bf(float f) {
    unsigned int u = __float_as_uint(f);
    return (unsigned short)((u + 0x7FFFu + ((u >> 16) & 1u)) >> 16);
}

// ---------------------------------------------------------------------------
// Kernel 0: one-time weight prep. Transpose + bf16-convert each 128x128 W
// into wt[which][outcol][inch] so projection B-fragments load contiguously.
// ---------------------------------------------------------------------------
__global__ void wprep_kernel(const float* __restrict__ Wq,
                             const float* __restrict__ Wk,
                             const float* __restrict__ Wv,
                             unsigned short* __restrict__ wt) {
    const int which = blockIdx.x;
    const float* W = (which == 0) ? Wq : (which == 1) ? Wk : Wv;
    unsigned short* o = wt + which * 128 * 128;
    for (int idx = threadIdx.x; idx < 128 * 128; idx += blockDim.x) {
        const int oc = idx >> 7;        // out column
        const int ic = idx & 127;       // in channel
        o[idx] = f2bf(W[ic * 128 + oc]);
    }
}

// ---------------------------------------------------------------------------
// Kernel 1: projections.  blockIdx.y = 0:q  1:k  2:v.
// 8 waves/block, each wave: 16 rows x 128 cols via 8x(4 WMMA) with K=32 chunks.
//   q -> qw[row*128 + h*32+d]  * 1/sqrt(32)   (row-major, frag friendly)
//   k -> kw[(h*L + row)*32 + d]               (row-major per head)
//   v -> vw[(h*32 + d)*L + row]               (transposed per head)
// ---------------------------------------------------------------------------
__global__ __launch_bounds__(256, 2) void proj_kernel(
    const float* __restrict__ qs, const float* __restrict__ ks, const float* __restrict__ vs,
    const unsigned short* __restrict__ wt,
    const float* __restrict__ bq, const float* __restrict__ bk, const float* __restrict__ bv,
    unsigned short* __restrict__ qw, unsigned short* __restrict__ kw,
    unsigned short* __restrict__ vw)
{
    const int which = blockIdx.y;
    const float* X = (which == 0) ? qs : (which == 1) ? ks : vs;
    const float* B = (which == 0) ? bq : (which == 1) ? bk : bv;
    const unsigned short* Wt = wt + which * 128 * 128;   // [outcol][inch] bf16
    const float oscale = (which == 0) ? 0.17677669529663687f : 1.0f; // fold 1/sqrt(dk) into q

    const int lane = threadIdx.x & 31;
    const int wave = threadIdx.x >> 5;
    const int m    = lane & 15;        // row (A) / col (B,C) within tile
    const int kh   = lane >> 4;        // K-half select
    const int row0 = blockIdx.x * 128 + wave * 16;
    const int arow = row0 + m;

    // A fragments of the input rows, one per 32-wide K chunk (f32 -> bf16)
    Frag16 A[4];
#pragma unroll
    for (int kc = 0; kc < 4; ++kc) {
        const float* p = X + arow * 128 + kc * 32 + kh * 8;
        f32x4 a0 = *(const f32x4*)(p);
        f32x4 a1 = *(const f32x4*)(p + 4);
        f32x4 a2 = *(const f32x4*)(p + 16);
        f32x4 a3 = *(const f32x4*)(p + 20);
        A[kc].u[0] = pk_bf16(a0.x, a0.y); A[kc].u[1] = pk_bf16(a0.z, a0.w);
        A[kc].u[2] = pk_bf16(a1.x, a1.y); A[kc].u[3] = pk_bf16(a1.z, a1.w);
        A[kc].u[4] = pk_bf16(a2.x, a2.y); A[kc].u[5] = pk_bf16(a2.z, a2.w);
        A[kc].u[6] = pk_bf16(a3.x, a3.y); A[kc].u[7] = pk_bf16(a3.z, a3.w);
    }

#pragma unroll
    for (int nc = 0; nc < 8; ++nc) {               // 8 output column tiles of 16
        const int col = nc * 16 + m;
        v8f acc = {};
#pragma unroll
        for (int kc = 0; kc < 4; ++kc) {
            // B-frag: lane holds out-column `col`, K values 8kh..+7 and 16+8kh..+7.
            Frag16 Bf;
            const unsigned short* p = Wt + col * 128 + kc * 32 + kh * 8;
            Bf.q[0] = *(const u32x4*)p;
            Bf.q[1] = *(const u32x4*)(p + 16);
            acc = __builtin_amdgcn_wmma_f32_16x16x32_bf16(
                false, A[kc].v, false, Bf.v, (short)0, acc, false, false);
        }
        const float bias = B[col];

        // ---- branchless store: all three layouts are base + row*rmul ------
        const int h  = nc >> 1;
        const int d0 = (nc & 1) * 16 + m;
        unsigned short* outp;
        size_t base;
        int rmul;
        if (which == 0)      { outp = qw; base = (size_t)col;                rmul = 128; }
        else if (which == 1) { outp = kw; base = (size_t)h * LL * 32 + d0;   rmul = 32;  }
        else                 { outp = vw; base = ((size_t)h * 32 + d0) * LL; rmul = 1;   }
#pragma unroll
        for (int r = 0; r < 8; ++r) {
            const int row = row0 + r + 8 * kh;
            outp[base + (size_t)row * rmul] = f2bf((acc[r] + bias) * oscale);
        }
    }
}

// ---------------------------------------------------------------------------
// Kernel 2: fused attention.  32 blocks x 256 threads (8 waves).
// Each wave owns 32 query rows (two 16-row n-chunks); L looped in 32-tiles.
// ---------------------------------------------------------------------------
__global__ __launch_bounds__(256, 1) void attn_kernel(
    const unsigned short* __restrict__ qws, const unsigned short* __restrict__ kws,
    const unsigned short* __restrict__ vws, float* __restrict__ out)
{
    const int lane = threadIdx.x & 31;
    const int wave = threadIdx.x >> 5;
    const int nl   = lane & 15;
    const int kh   = lane >> 4;
    const bool hi  = (kh != 0);
    const int n0   = blockIdx.x * 256 + wave * 32;

    // Q as B-fragments (K^T x Q^T): lane holds column n, dk 8kh..+7 and 16+8kh..+7.
    Frag16 Qf[4][2];
#pragma unroll
    for (int h = 0; h < 4; ++h)
#pragma unroll
        for (int c = 0; c < 2; ++c) {
            const unsigned short* p = qws + (size_t)(n0 + c * 16 + nl) * 128 + h * 32 + kh * 8;
            Qf[h][c].q[0] = *(const u32x4*)p;
            Qf[h][c].q[1] = *(const u32x4*)(p + 16);
        }

    v8f O[4][2][2] = {};                 // O^T accumulators [h][dchunk][nchunk]

    for (int lt = 0; lt < LL / 32; ++lt) {
        const int l0 = lt * 32;

        // ---- scores (transposed, pre-scaled): S^T[h][j][c] = Ktile x Q^T --
        v8f S[4][2][2] = {};
#pragma unroll
        for (int h = 0; h < 4; ++h)
#pragma unroll
            for (int j = 0; j < 2; ++j) {
                Frag16 Kf;               // A-frag: rows l, K=dk (row-major)
                const unsigned short* p =
                    kws + ((size_t)h * LL + l0 + j * 16 + nl) * 32 + kh * 8;
                Kf.q[0] = *(const u32x4*)p;
                Kf.q[1] = *(const u32x4*)(p + 16);
#pragma unroll
                for (int c = 0; c < 2; ++c)
                    S[h][j][c] = __builtin_amdgcn_wmma_f32_16x16x32_bf16(
                        false, Kf.v, false, Qf[h][c].v, (short)0, S[h][j][c], false, false);
            }

        // ---- softmax over the 4 heads, elementwise per (n,l) slot ---------
#pragma unroll
        for (int j = 0; j < 2; ++j)
#pragma unroll
            for (int c = 0; c < 2; ++c)
#pragma unroll
                for (int r = 0; r < 8; ++r) {
                    float t0 = S[0][j][c][r], t1 = S[1][j][c][r];
                    float t2 = S[2][j][c][r], t3 = S[3][j][c][r];
                    float mx = fmaxf(fmaxf(t0, t1), fmaxf(t2, t3));
                    float e0 = __expf(t0 - mx), e1 = __expf(t1 - mx);
                    float e2 = __expf(t2 - mx), e3 = __expf(t3 - mx);
                    float inv = __builtin_amdgcn_rcpf(e0 + e1 + e2 + e3);
                    S[0][j][c][r] = e0 * inv; S[1][j][c][r] = e1 * inv;
                    S[2][j][c][r] = e2 * inv; S[3][j][c][r] = e3 * inv;
                }

        // ---- O^T += V^T x W^T  (W^T B-frag via SWAPX16 lane swap) ---------
#pragma unroll
        for (int h = 0; h < 4; ++h) {
            Frag16 Vf[2];                // A-frags: rows d, K=l (v stored [h][d][L])
#pragma unroll
            for (int dd = 0; dd < 2; ++dd) {
                const unsigned short* p =
                    vws + ((size_t)h * 32 + dd * 16 + nl) * LL + l0 + kh * 16;
                Vf[dd].q[0] = *(const u32x4*)p;
                Vf[dd].q[1] = *(const u32x4*)(p + 8);
            }
#pragma unroll
            for (int c = 0; c < 2; ++c) {
                unsigned p0[4], p1[4];
#pragma unroll
                for (int mm = 0; mm < 4; ++mm) {
                    p0[mm] = pk_bf16(S[h][0][c][2 * mm], S[h][0][c][2 * mm + 1]);
                    p1[mm] = pk_bf16(S[h][1][c][2 * mm], S[h][1][c][2 * mm + 1]);
                }
                Frag16 Bw;               // B-frag of W^T[32l x 16n]
#pragma unroll
                for (int mm = 0; mm < 4; ++mm) {
                    unsigned s0 = (unsigned)__builtin_amdgcn_ds_swizzle((int)p0[mm], 0x401F); // SWAPX16
                    unsigned s1 = (unsigned)__builtin_amdgcn_ds_swizzle((int)p1[mm], 0x401F);
                    Bw.u[mm]     = hi ? s1     : p0[mm];
                    Bw.u[mm + 4] = hi ? p1[mm] : s0;
                }
#pragma unroll
                for (int dd = 0; dd < 2; ++dd)
                    O[h][dd][c] = __builtin_amdgcn_wmma_f32_16x16x32_bf16(
                        false, Vf[dd].v, false, Bw.v, (short)0, O[h][dd][c], false, false);
            }
        }

        // ---- prefetch upcoming K/V tiles (L2-resident stream) -------------
        if (lt + 4 < LL / 32) {
            __builtin_prefetch(kws + (size_t)(l0 + 128 + nl) * 32, 0, 0);
            __builtin_prefetch(vws + (size_t)nl * LL + l0 + 128, 0, 0);
        }
    }

    // ---- write out[n,h,d] from O^T C-layout (col n per lane, rows d) ------
#pragma unroll
    for (int h = 0; h < 4; ++h)
#pragma unroll
        for (int dd = 0; dd < 2; ++dd)
#pragma unroll
            for (int c = 0; c < 2; ++c) {
                const int ng = n0 + c * 16 + nl;
#pragma unroll
                for (int r = 0; r < 8; ++r) {
                    const int d = dd * 16 + r + 8 * kh;
                    out[((size_t)ng * 4 + h) * 32 + d] = O[h][dd][c][r];
                }
            }
}

// ---------------------------------------------------------------------------
extern "C" void kernel_launch(void* const* d_in, const int* in_sizes, int n_in,
                              void* d_out, int out_size, void* d_ws, size_t ws_size,
                              hipStream_t stream) {
    (void)in_sizes; (void)n_in; (void)out_size; (void)ws_size;
    const float* qs = (const float*)d_in[0];
    const float* ks = (const float*)d_in[1];
    const float* vs = (const float*)d_in[2];
    const float* Wq = (const float*)d_in[3];
    const float* bq = (const float*)d_in[4];
    const float* Wk = (const float*)d_in[5];
    const float* bk = (const float*)d_in[6];
    const float* Wv = (const float*)d_in[7];
    const float* bv = (const float*)d_in[8];

    unsigned short* ws = (unsigned short*)d_ws;
    unsigned short* qw = ws;                          // [8192][128] bf16 (2 MB)
    unsigned short* kw = ws + (size_t)NN * 128;       // [4][8192][32] bf16 (2 MB)
    unsigned short* vw = ws + (size_t)2 * NN * 128;   // [4][32][8192] bf16 (2 MB)
    unsigned short* wt = ws + (size_t)3 * NN * 128;   // [3][128][128] bf16 (96 KB)

    wprep_kernel<<<dim3(3), 256, 0, stream>>>(Wq, Wk, Wv, wt);
    proj_kernel<<<dim3(NN / 128, 3), 256, 0, stream>>>(
        qs, ks, vs, wt, bq, bk, bv, qw, kw, vw);
    attn_kernel<<<dim3(NN / 256), 256, 0, stream>>>(qw, kw, vw, (float*)d_out);
}
